// Projector_5892695130630
// MI455X (gfx1250) — matrix operand order repaired and checked by reading, not compiled, not run on hardware
//
#include <hip/hip_runtime.h>
#include <hip/hip_bf16.h>
#include <math.h>

// Problem constants (from reference)
#define BSZ   8
#define CLSN  4
#define CCH   256
#define HH    128
#define WWD   128
#define HWN   (HH*WWD)     // 16384
#define LTOK  19
#define TKD   256
#define NPIX  256          // pixels per workgroup

typedef __attribute__((ext_vector_type(8)))  float  v8f;
typedef __attribute__((ext_vector_type(16))) __bf16 v16bf;

union Frag { v16bf v; unsigned int u[8]; };

__device__ __forceinline__ unsigned int packbf2(float a, float b) {
  unsigned int ua = __float_as_uint(a); ua = (ua + 0x7FFFu + ((ua >> 16) & 1u)) >> 16;
  unsigned int ub = __float_as_uint(b); ub = (ub + 0x7FFFu + ((ub >> 16) & 1u)) >> 16;
  return ua | (ub << 16);
}
__device__ __forceinline__ unsigned short f2bf(float a) {
  unsigned int ua = __float_as_uint(a); ua = (ua + 0x7FFFu + ((ua >> 16) & 1u)) >> 16;
  return (unsigned short)ua;
}

// ---------------------------------------------------------------------------
// Precompute per batch:
//   t_q[l][t] = sum_c token_s[l][c] * W2[t][c]
//   M[l][c]   = sum_t t_q[l][t]    * W1[t][c]   (bf16, l padded to 32 w/ zeros)
//   t3T[o][l] = sum_t token_s[l][t]* W3[o][t] + b3[o]  (bf16, transposed, l pad 32)
// ---------------------------------------------------------------------------
__global__ __launch_bounds__(256) void precompute_kernel(
    const float* __restrict__ token_s, const float* __restrict__ W1,
    const float* __restrict__ W2, const float* __restrict__ W3,
    const float* __restrict__ b3,
    unsigned short* __restrict__ Mbf, unsigned short* __restrict__ t3T) {
  __shared__ float ts[LTOK * TKD];
  __shared__ float tq[LTOK * TKD];
  const int b = blockIdx.x, tid = threadIdx.x;
  for (int i = tid; i < LTOK * TKD; i += 256) ts[i] = token_s[b * LTOK * TKD + i];
  __syncthreads();
  {  // t_q: thread tid owns column t=tid
    const float* w2 = W2 + (size_t)tid * TKD;
    for (int l = 0; l < LTOK; ++l) {
      float acc = 0.f;
      for (int c = 0; c < TKD; ++c) acc += ts[l * TKD + c] * w2[c];
      tq[l * TKD + tid] = acc;
    }
  }
  __syncthreads();
  {  // M: thread tid owns column c=tid
    for (int l = 0; l < 32; ++l) {
      float acc = 0.f;
      if (l < LTOK)
        for (int t = 0; t < TKD; ++t) acc += tq[l * TKD + t] * W1[(size_t)t * CCH + tid];
      Mbf[(size_t)b * 32 * CCH + l * CCH + tid] = f2bf(acc);
    }
  }
  {  // t3T: thread tid owns output channel o=tid
    const float* w3 = W3 + (size_t)tid * TKD;
    for (int l = 0; l < 32; ++l) {
      float acc = 0.f;
      if (l < LTOK) {
        for (int t = 0; t < TKD; ++t) acc += ts[l * TKD + t] * w3[t];
        acc += b3[tid];
      }
      t3T[(size_t)b * CCH * 32 + tid * 32 + l] = f2bf(acc);
    }
  }
}

// ---------------------------------------------------------------------------
// Main fused kernel: mask-weighted feature -> GEMM1 (q x M^T) -> softmax ->
// GEMM2 (a_soft x t3) -> +fea_t residual -> store. One block = (batch, 256 px).
// ---------------------------------------------------------------------------
__global__ __launch_bounds__(256) void proj_main_kernel(
    const float* __restrict__ mask_t, const float* __restrict__ fea,
    const unsigned int* __restrict__ Mbf32, const unsigned int* __restrict__ t3T32,
    float* __restrict__ out) {
  // LDS: row padded to 17/129 dwords for bank-conflict-free access
  __shared__ unsigned int MA[32 * 129];   // M, bf16 pairs, row = 256 ch
  __shared__ unsigned int QB[NPIX * 17];  // q chunk, bf16 pairs, row = 32 ch
  __shared__ unsigned int T3[256 * 17];   // t3T, bf16 pairs, row = 32 l

  const int tid = threadIdx.x;
  const int b = blockIdx.x >> 6;
  const int pixbase = (blockIdx.x & 63) * NPIX;

  for (int i = tid; i < 4096; i += 256) MA[(i >> 7) * 129 + (i & 127)] = Mbf32[b * 4096 + i];
  for (int i = tid; i < 4096; i += 256) T3[(i >> 4) * 17 + (i & 15)] = t3T32[b * 4096 + i];

  // Per-pixel mask weights: nearest upsample 64->128 + mean over cls (x0.25)
  const int p = pixbase + tid;
  const int h = p >> 7, w = p & 127;
  const int mb = (b * CLSN) * 4096 + (h >> 1) * 64 + (w >> 1);
  const float mw0 = 0.25f * mask_t[mb];
  const float mw1 = 0.25f * mask_t[mb + 4096];
  const float mw2 = 0.25f * mask_t[mb + 2 * 4096];
  const float mw3 = 0.25f * mask_t[mb + 3 * 4096];
  // fea source batches: (4b+cl) % 8 = 4*(b&1)+cl
  const int sb = (b & 1) * 4;
  const float* f0 = fea + (size_t)(sb + 0) * CCH * HWN + p;
  const float* f1 = fea + (size_t)(sb + 1) * CCH * HWN + p;
  const float* f2 = fea + (size_t)(sb + 2) * CCH * HWN + p;
  const float* f3 = fea + (size_t)(sb + 3) * CCH * HWN + p;
  __syncthreads();

  const int lane = tid & 31;
  const int wbase = (tid >> 5) * 32;  // 32 pixels per wave
  const bool lo = lane < 16;
  const int l15 = lane & 15;

  v8f acc[2][2];  // [l-tile][pixel-tile]
#pragma unroll
  for (int lt = 0; lt < 2; ++lt)
#pragma unroll
    for (int nt = 0; nt < 2; ++nt)
#pragma unroll
      for (int r = 0; r < 8; ++r) acc[lt][nt][r] = 0.f;

  // ---- GEMM1: a[l, pix] += M[l, c] * q[c, pix], K-chunks of 32 channels ----
  for (int kc = 0; kc < 8; ++kc) {
    unsigned int* qrow = &QB[tid * 17];
#pragma unroll
    for (int j = 0; j < 16; ++j) {
      const size_t c0 = (size_t)(kc * 32 + 2 * j) * HWN;
      const size_t c1 = c0 + HWN;
      float q0 = mw0 * f0[c0] + mw1 * f1[c0] + mw2 * f2[c0] + mw3 * f3[c0];
      float q1 = mw0 * f0[c1] + mw1 * f1[c1] + mw2 * f2[c1] + mw3 * f3[c1];
      qrow[j] = packbf2(q0, q1);
    }
    __syncthreads();

    Frag afr[2], bfr[2];
    const int aoff = lo ? 0 : 4;
#pragma unroll
    for (int lt = 0; lt < 2; ++lt) {
      const unsigned int* src = &MA[(lt * 16 + l15) * 129 + kc * 16 + aoff];
#pragma unroll
      for (int r = 0; r < 8; ++r) afr[lt].u[r] = src[(r < 4) ? r : r + 4];
    }
    const int boff = lo ? 0 : 4;
#pragma unroll
    for (int nt = 0; nt < 2; ++nt) {
      const unsigned int* src = &QB[(wbase + nt * 16 + l15) * 17 + boff];
#pragma unroll
      for (int r = 0; r < 8; ++r) bfr[nt].u[r] = src[(r < 4) ? r : r + 4];
    }
#pragma unroll
    for (int lt = 0; lt < 2; ++lt)
#pragma unroll
      for (int nt = 0; nt < 2; ++nt)
        acc[lt][nt] = __builtin_amdgcn_wmma_f32_16x16x32_bf16(
            false, afr[lt].v, false, bfr[nt].v, (short)0, acc[lt][nt], false, false);
    __syncthreads();
  }

  // ---- softmax over l (19 valid), zero where a==0; build GEMM2 B-frags ----
  // Lane p   (<16): holds l=0..7 (acc[0]) and l=16..23 (acc[1]) for pixel p
  // Lane p+16     : holds l=8..15 (acc[0]) and l=24..31 (acc[1], all invalid)
  unsigned int bq[2][8];
  const float s = 0.0625f;  // 1/sqrt(256)
#pragma unroll
  for (int nt = 0; nt < 2; ++nt) {
    float x0[8], x1[3];
#pragma unroll
    for (int r = 0; r < 8; ++r) x0[r] = acc[0][nt][r] * s;
#pragma unroll
    for (int r = 0; r < 3; ++r) x1[r] = acc[1][nt][r] * s;
    float m = x0[0];
#pragma unroll
    for (int r = 1; r < 8; ++r) m = fmaxf(m, x0[r]);
    if (lo) {
#pragma unroll
      for (int r = 0; r < 3; ++r) m = fmaxf(m, x1[r]);
    }
    m = fmaxf(m, __shfl_xor(m, 16, 32));
    float e0[8], e1[3], sum = 0.f;
#pragma unroll
    for (int r = 0; r < 8; ++r) { e0[r] = __expf(x0[r] - m); sum += e0[r]; }
#pragma unroll
    for (int r = 0; r < 3; ++r) { e1[r] = lo ? __expf(x1[r] - m) : 0.f; sum += e1[r]; }
    sum += __shfl_xor(sum, 16, 32);
    const float inv = 1.f / sum;
    float p0[8], p1[8];
#pragma unroll
    for (int r = 0; r < 8; ++r) p0[r] = (acc[0][nt][r] != 0.f) ? e0[r] * inv : 0.f;
#pragma unroll
    for (int r = 0; r < 8; ++r)
      p1[r] = (lo && r < 3 && acc[1][nt][r] != 0.f) ? e1[r] * inv : 0.f;
    unsigned int P0[4], P1[4];
#pragma unroll
    for (int j = 0; j < 4; ++j) {
      P0[j] = packbf2(p0[2 * j], p0[2 * j + 1]);
      P1[j] = packbf2(p1[2 * j], p1[2 * j + 1]);
    }
    // B-fragment of a_soft [K=32 l x 16 pix]: exchange halves across lane^16
#pragma unroll
    for (int j = 0; j < 4; ++j) {
      const unsigned int o0 = __shfl_xor(P0[j], 16, 32);
      const unsigned int o1 = __shfl_xor(P1[j], 16, 32);
      bq[nt][j]     = lo ? P0[j] : o1;  // l 0..7   / 16..23
      bq[nt][4 + j] = lo ? o0 : P1[j];  // l 8..15  / 24..31
    }
  }

  // ---- GEMM2: out[o, pix] = t3T[o, l] * a_soft[l, pix]; +fea residual ----
  const float* feab = fea + (size_t)b * CCH * HWN;
  float* outb = out + (size_t)b * CCH * HWN;
  const int aoff2 = lo ? 0 : 8;
  for (int ot = 0; ot < 16; ++ot) {
    Frag t3f;
    const unsigned int* src = &T3[(ot * 16 + l15) * 17 + aoff2];
#pragma unroll
    for (int r = 0; r < 8; ++r) t3f.u[r] = src[r];
#pragma unroll
    for (int nt = 0; nt < 2; ++nt) {
      Frag bf;
#pragma unroll
      for (int r = 0; r < 8; ++r) bf.u[r] = bq[nt][r];
      v8f z;
#pragma unroll
      for (int r = 0; r < 8; ++r) z[r] = 0.f;
      v8f d = __builtin_amdgcn_wmma_f32_16x16x32_bf16(
          false, t3f.v, false, bf.v, (short)0, z, false, false);
      const int pix = pixbase + wbase + nt * 16 + l15;
      const int obase = ot * 16 + (lo ? 0 : 8);
#pragma unroll
      for (int r = 0; r < 8; ++r) {
        const size_t idx = (size_t)(obase + r) * HWN + pix;
        outb[idx] = d[r] + feab[idx];
      }
    }
  }
}

extern "C" void kernel_launch(void* const* d_in, const int* in_sizes, int n_in,
                              void* d_out, int out_size, void* d_ws, size_t ws_size,
                              hipStream_t stream) {
  (void)in_sizes; (void)n_in; (void)out_size; (void)ws_size;
  const float* mask_t  = (const float*)d_in[0];
  const float* fea_t   = (const float*)d_in[1];
  const float* token_s = (const float*)d_in[2];
  const float* W1 = (const float*)d_in[3];
  const float* W2 = (const float*)d_in[4];
  const float* W3 = (const float*)d_in[5];
  const float* b3 = (const float*)d_in[6];
  float* out = (float*)d_out;

  unsigned short* Mbf = (unsigned short*)d_ws;                       // 8*32*256*2 = 128 KiB
  unsigned short* t3T = (unsigned short*)((char*)d_ws + 8 * 32 * 256 * 2);  // 128 KiB

  precompute_kernel<<<8, 256, 0, stream>>>(token_s, W1, W2, W3, b3, Mbf, t3T);
  proj_main_kernel<<<BSZ * (HWN / NPIX), 256, 0, stream>>>(
      mask_t, fea_t, (const unsigned int*)Mbf, (const unsigned int*)t3T, out);
}
